// VTV_GCL_18580028522829
// MI455X (gfx1250) — compile-verified, matrix operand order checked by reading, not run on hardware
//
#include <hip/hip_runtime.h>
#include <hip/hip_bf16.h>
#include <cstddef>

// ---------------------------------------------------------------------------
// Problem constants (static per reference)
// ---------------------------------------------------------------------------
#define N_NODES 512
#define E_EDGES 512
#define M_NB    16384
#define FIN     64
#define HID     128
#define DPE     32
#define EN      512
#define R_TOTAL (EN * EN)          // 262144 flattened (i,j) rows of T

typedef __bf16 bf16;
typedef __attribute__((ext_vector_type(16))) __bf16 v16bf;
typedef __attribute__((ext_vector_type(8)))  __bf16 v8bf;
typedef __attribute__((ext_vector_type(8)))  float  v8f;

union U16 { v16bf v; v8bf h[2]; };

// ---------------------------------------------------------------------------
// WMMA helpers (CDNA5 v_wmma_f32_16x16x32_bf16, wave32)
// ---------------------------------------------------------------------------
__device__ __forceinline__ v8f wmma_bf16(v16bf a, v16bf b, v8f c) {
  return __builtin_amdgcn_wmma_f32_16x16x32_bf16(false, a, false, b, (short)0, c,
                                                 false, false);
}

// Fragment load from a row-major tile (leading dim ld, elements).
// Used for BOTH A (M-major, 16xK) and B (N-major, 16xK transposed) operands:
// lane & 15 selects the row (M or N), lane >> 4 selects k 0-7,16-23 (lanes 0-15) or 8-15,24-31 (lanes 16-31).
// Two contiguous 16-byte LDS loads per fragment.
__device__ __forceinline__ v16bf load_frag(const bf16* src, int ld) {
  int lane = threadIdx.x & 31;
  const bf16* p = src + (lane & 15) * ld + ((lane >> 4) << 3);
  U16 u;
  u.h[0] = *(const v8bf*)(p);
  u.h[1] = *(const v8bf*)(p + 16);
  return u.v;
}

__device__ __forceinline__ float silu_f(float x) {
  return x * (1.0f / (1.0f + __expf(-x)));
}

// ---------------------------------------------------------------------------
// Small utility kernels
// ---------------------------------------------------------------------------
__global__ void k_zero_f32(float* p, int n) {
  for (int i = blockIdx.x * blockDim.x + threadIdx.x; i < n;
       i += gridDim.x * blockDim.x)
    p[i] = 0.0f;
}

__global__ void k_f2b(const float* __restrict__ s, bf16* __restrict__ d, int n) {
  for (int i = blockIdx.x * blockDim.x + threadIdx.x; i < n;
       i += gridDim.x * blockDim.x)
    d[i] = (bf16)s[i];
}

// fp32 [K][N] -> bf16 transposed [N][K]
__global__ void k_f2b_t(const float* __restrict__ s, bf16* __restrict__ d, int K,
                        int N) {
  for (int i = blockIdx.x * blockDim.x + threadIdx.x; i < K * N;
       i += gridDim.x * blockDim.x) {
    int n = i / K, k = i - n * K;
    d[i] = (bf16)s[k * N + n];
  }
}

// coord_diff + concat(h[rows], h[cols]) as bf16
__global__ void k_prep(const float* __restrict__ h, const float* __restrict__ x,
                       const int* __restrict__ edges, float* __restrict__ cd,
                       bf16* __restrict__ xh) {
  int e = blockIdx.x * blockDim.x + threadIdx.x;
  if (e >= E_EDGES) return;
  int r = edges[e], c = edges[E_EDGES + e];
#pragma unroll
  for (int d = 0; d < 3; ++d) cd[e * 3 + d] = x[r * 3 + d] - x[c * 3 + d];
  for (int f = 0; f < FIN; ++f) {
    xh[e * 2 * FIN + f]       = (bf16)h[r * FIN + f];
    xh[e * 2 * FIN + FIN + f] = (bf16)h[c * FIN + f];
  }
}

// vtv + sinusoidal PE + efn-product, scatter-add into dense T[512,512,96]
__global__ void k_vtv_scatter(const int* __restrict__ nb_edge,
                              const float* __restrict__ cd,
                              const bf16* __restrict__ efn,
                              float* __restrict__ T) {
  int m = blockIdx.x;
  int c = threadIdx.x;  // 0..95
  int a = nb_edge[m], b = nb_edge[M_NB + m];
  float vtv = cd[a * 3 + 0] * cd[b * 3 + 0] + cd[a * 3 + 1] * cd[b * 3 + 1] +
              cd[a * 3 + 2] * cd[b * 3 + 2];
  float val;
  if (c < DPE) {
    int half = c >> 1;
    float ang = vtv * 16.0f * __expf(-0.5756462732f * (float)half);
    val = (c & 1) ? __cosf(ang) : __sinf(ang);
  } else {
    int f = c - DPE;
    val = (float)efn[a * FIN + f] * (float)efn[b * FIN + f];
  }
  atomicAdd(&T[((size_t)a * EN + (size_t)b) * 96 + c], val);
}

// ---------------------------------------------------------------------------
// Fused 2-layer MLP via WMMA:  out = SiLU(X @ W1 + b1) @ W2 + b2
// W1t/W2t are TRANSPOSED (N-major) bf16 so B fragments are contiguous loads.
// X = [rowMajor X0 (K0 cols) | feature-major Xcol (KC planes)].
// 128 threads (4 waves), 64 rows per workgroup (16 rows per wave).
// fmT: write feature-major output with (i,j)->(j,i) transposed row index.
// ---------------------------------------------------------------------------
__global__ void k_fused_mlp(const bf16* __restrict__ X0, int K0,
                            const bf16* __restrict__ Xcol, int KC,
                            const bf16* __restrict__ W1t, const float* __restrict__ B1,
                            const bf16* __restrict__ W2t, const float* __restrict__ B2,
                            int rowsTotal,
                            bf16* __restrict__ outRM,
                            bf16* __restrict__ outFM, int fmT,
                            float* __restrict__ outF32) {
  const int K1 = K0 + KC;
  extern __shared__ char smem_raw[];
  bf16* sW1 = (bf16*)smem_raw;         // [HID][K1]   (N-major)
  bf16* sW2 = sW1 + K1 * HID;          // [FIN][HID]  (N-major)
  bf16* sX  = sW2 + HID * FIN;         // [64][K1]
  bf16* sH  = sX + 64 * K1;            // [64][HID]

  int tid = threadIdx.x;
  int r0  = blockIdx.x * 64;

  // stage weights (contiguous 16B copies)
  for (int i = tid; i < (K1 * HID) >> 3; i += blockDim.x)
    ((v8bf*)sW1)[i] = ((const v8bf*)W1t)[i];
  for (int i = tid; i < (HID * FIN) >> 3; i += blockDim.x)
    ((v8bf*)sW2)[i] = ((const v8bf*)W2t)[i];
  // stage X tile
  {
    int K0v = K0 >> 3;
    for (int i = tid; i < 64 * K0v; i += blockDim.x) {
      int r = i / K0v, c = i - r * K0v;
      *(v8bf*)(sX + r * K1 + c * 8) =
          *(const v8bf*)(X0 + (size_t)(r0 + r) * K0 + c * 8);
    }
  }
  if (KC > 0) {
    for (int i = tid; i < 64 * KC; i += blockDim.x) {
      int c = i >> 6, r = i & 63;  // coalesced along rows within a feature plane
      sX[r * K1 + K0 + c] = Xcol[(size_t)c * (size_t)rowsTotal + r0 + r];
    }
  }
  __syncthreads();

  int lane  = tid & 31;
  int wave  = tid >> 5;
  int wr0   = wave * 16;
  int nloc  = lane & 15;
  int mbase = (lane >> 4) * 8;

  // ---- layer 1: H(16x128) = X(16xK1) @ W1(K1x128) + b1 ----
  v8f acc1[8];
#pragma unroll
  for (int nt = 0; nt < 8; ++nt) {
    float bv = B1[nt * 16 + nloc];
#pragma unroll
    for (int v = 0; v < 8; ++v) acc1[nt][v] = bv;
  }
  for (int k = 0; k < K1; k += 32) {
    v16bf a = load_frag(sX + wr0 * K1 + k, K1);
#pragma unroll
    for (int nt = 0; nt < 8; ++nt) {
      v16bf b = load_frag(sW1 + (nt * 16) * K1 + k, K1);
      acc1[nt] = wmma_bf16(a, b, acc1[nt]);
    }
  }
  // SiLU, C-layout -> A-layout through LDS
#pragma unroll
  for (int nt = 0; nt < 8; ++nt)
#pragma unroll
    for (int v = 0; v < 8; ++v)
      sH[(wr0 + mbase + v) * HID + nt * 16 + nloc] = (bf16)silu_f(acc1[nt][v]);
  __syncthreads();

  // ---- layer 2: OUT(16x64) = H(16x128) @ W2(128x64) + b2 ----
  v8f acc2[4];
#pragma unroll
  for (int nt = 0; nt < 4; ++nt) {
    float bv = B2[nt * 16 + nloc];
#pragma unroll
    for (int v = 0; v < 8; ++v) acc2[nt][v] = bv;
  }
#pragma unroll
  for (int k = 0; k < HID; k += 32) {
    v16bf a = load_frag(sH + wr0 * HID + k, HID);
#pragma unroll
    for (int nt = 0; nt < 4; ++nt) {
      v16bf b = load_frag(sW2 + (nt * 16) * HID + k, HID);
      acc2[nt] = wmma_bf16(a, b, acc2[nt]);
    }
  }

  if (outRM) {
#pragma unroll
    for (int nt = 0; nt < 4; ++nt)
#pragma unroll
      for (int v = 0; v < 8; ++v)
        outRM[(size_t)(r0 + wr0 + mbase + v) * FIN + nt * 16 + nloc] =
            (bf16)acc2[nt][v];
  }
  if (outFM) {
    if (!fmT) {
      // feature-major: the 8 accumulator rows per lane are CONTIGUOUS -> 16B store
#pragma unroll
      for (int nt = 0; nt < 4; ++nt) {
        v8bf pk;
#pragma unroll
        for (int v = 0; v < 8; ++v) pk[v] = (bf16)acc2[nt][v];
        *(v8bf*)(outFM + (size_t)(nt * 16 + nloc) * (size_t)rowsTotal + r0 + wr0 +
                 mbase) = pk;
      }
    } else {
      // transposed plane [j][k] for the einsum B operand
#pragma unroll
      for (int nt = 0; nt < 4; ++nt)
#pragma unroll
        for (int v = 0; v < 8; ++v) {
          int row = r0 + wr0 + mbase + v;             // row = k*512 + j
          int tr  = ((row & (EN - 1)) << 9) | (row >> 9);  // -> j*512 + k
          outFM[(size_t)(nt * 16 + nloc) * (size_t)rowsTotal + tr] =
              (bf16)acc2[nt][v];
        }
    }
  }
  if (outF32) {
    // stage fp32 tile in LDS (reuse sH: same 16KB footprint), coalesced store
    float* sO = (float*)sH;
#pragma unroll
    for (int nt = 0; nt < 4; ++nt)
#pragma unroll
      for (int v = 0; v < 8; ++v)
        sO[(wr0 + mbase + v) * FIN + nt * 16 + nloc] = acc2[nt][v];
    __syncthreads();
    for (int i = tid; i < (64 * FIN) >> 2; i += blockDim.x)
      *(float4*)(outF32 + (size_t)r0 * FIN + i * 4) = *(float4*)(sO + i * 4);
  }
}

// ---------------------------------------------------------------------------
// Per-feature batched GEMM: mult_f(512x512) = m1_f(512x512) @ m2_f(512x512)
// A planes row-major [i][k]; B planes TRANSPOSED [j][k] (N-major).
// 256 threads, 128x128 C tile, K steps of 32; all frags are 2x16B LDS loads.
// ---------------------------------------------------------------------------
__global__ void k_einsum(const bf16* __restrict__ Aall,
                         const bf16* __restrict__ BTall,
                         bf16* __restrict__ Call) {
  int f    = blockIdx.y;
  int tile = blockIdx.x;               // 0..15
  int ti   = (tile >> 2) * 128;
  int tj   = (tile & 3) * 128;
  const bf16* A  = Aall  + (size_t)f * R_TOTAL;   // [i][k]
  const bf16* BT = BTall + (size_t)f * R_TOTAL;   // [j][k]
  bf16*       C  = Call  + (size_t)f * R_TOTAL;   // [i][j]

  __shared__ bf16 sA[128 * 32];   // [i][k]
  __shared__ bf16 sB[128 * 32];   // [j][k]
  __shared__ bf16 sC[128 * 128];

  int tid  = threadIdx.x;
  int wave = tid >> 5;
  int wr   = (wave >> 1) * 32;         // 0/32/64/96
  int wc   = (wave & 1) * 64;          // 0/64

  v8f acc[2][4];
#pragma unroll
  for (int i = 0; i < 2; ++i)
#pragma unroll
    for (int j = 0; j < 4; ++j)
#pragma unroll
      for (int v = 0; v < 8; ++v) acc[i][j][v] = 0.0f;

  for (int k0 = 0; k0 < EN; k0 += 32) {
    __syncthreads();
    // 512 16B chunks per tile, coalesced
    for (int i = tid; i < 512; i += 256) {
      int r = i >> 2, c = i & 3;
      *(v8bf*)(sA + r * 32 + c * 8) =
          *(const v8bf*)(A + (size_t)(ti + r) * EN + k0 + c * 8);
    }
    for (int i = tid; i < 512; i += 256) {
      int r = i >> 2, c = i & 3;
      *(v8bf*)(sB + r * 32 + c * 8) =
          *(const v8bf*)(BT + (size_t)(tj + r) * EN + k0 + c * 8);
    }
    if (k0 + 32 < EN) {  // warm caches for next K tile (global_prefetch)
      __builtin_prefetch(&A[(size_t)(ti + (tid >> 1)) * EN + k0 + 32], 0, 1);
      __builtin_prefetch(&BT[(size_t)(tj + (tid >> 1)) * EN + k0 + 32], 0, 1);
    }
    __syncthreads();
    v16bf a0 = load_frag(sA + (wr + 0) * 32, 32);
    v16bf a1 = load_frag(sA + (wr + 16) * 32, 32);
#pragma unroll
    for (int nt = 0; nt < 4; ++nt) {
      v16bf b = load_frag(sB + (wc + nt * 16) * 32, 32);
      acc[0][nt] = wmma_bf16(a0, b, acc[0][nt]);
      acc[1][nt] = wmma_bf16(a1, b, acc[1][nt]);
    }
  }

  // stage C tile in LDS, then coalesced 16B global stores
  int lane = tid & 31, nloc = lane & 15, mbase = (lane >> 4) * 8;
  __syncthreads();
#pragma unroll
  for (int rf = 0; rf < 2; ++rf)
#pragma unroll
    for (int nt = 0; nt < 4; ++nt)
#pragma unroll
      for (int v = 0; v < 8; ++v)
        sC[(wr + rf * 16 + mbase + v) * 128 + wc + nt * 16 + nloc] =
            (bf16)acc[rf][nt][v];
  __syncthreads();
  for (int i = tid; i < 2048; i += 256) {   // 128x128 bf16 = 2048 x 16B
    int r = i >> 4, c = i & 15;
    *(v8bf*)(C + (size_t)(ti + r) * EN + tj + c * 8) = *(v8bf*)(sC + r * 128 + c * 8);
  }
}

// ---------------------------------------------------------------------------
// IGN 2->1 reductions over Tout [512,512,64] fp32
// ---------------------------------------------------------------------------
__global__ void k_reduce_rows(const float* __restrict__ Tout, float* rowsum,
                              float* diag) {
  int i = blockIdx.x, f = threadIdx.x;
  float s = 0.0f;
  for (int j = 0; j < EN; ++j) s += Tout[((size_t)i * EN + j) * FIN + f];
  rowsum[i * FIN + f] = s;
  diag[i * FIN + f]   = Tout[((size_t)i * EN + i) * FIN + f];
}

__global__ void k_reduce_cols(const float* __restrict__ Tout, float* colsum) {
  int j = blockIdx.x, f = threadIdx.x;
  float s = 0.0f;
  for (int i = 0; i < EN; ++i) s += Tout[((size_t)i * EN + j) * FIN + f];
  colsum[j * FIN + f] = s;
}

__global__ void k_reduce_tot(const float* __restrict__ rowsum,
                             const float* __restrict__ diag, float* tot,
                             float* tr) {
  int f = threadIdx.x;
  float a = 0.0f, b = 0.0f;
  for (int i = 0; i < EN; ++i) {
    a += rowsum[i * FIN + f];
    b += diag[i * FIN + f];
  }
  tot[f] = a;
  tr[f]  = b;
}

// nb_t1 = concat(diag,rowsum,colsum,tot,tr)[E,320] @ ign_w + ign_b
__global__ void k_nb_t1(const float* __restrict__ diag, const float* __restrict__ rowsum,
                        const float* __restrict__ colsum, const float* __restrict__ tot,
                        const float* __restrict__ tr, const float* __restrict__ ign_w,
                        const float* __restrict__ ign_b, float* __restrict__ nb_t1) {
  int e = blockIdx.x, f = threadIdx.x;
  float s = ign_b[f];
  for (int c = 0; c < FIN; ++c) s += diag[e * FIN + c]   * ign_w[(0 * FIN + c) * FIN + f];
  for (int c = 0; c < FIN; ++c) s += rowsum[e * FIN + c] * ign_w[(1 * FIN + c) * FIN + f];
  for (int c = 0; c < FIN; ++c) s += colsum[e * FIN + c] * ign_w[(2 * FIN + c) * FIN + f];
  for (int c = 0; c < FIN; ++c) s += tot[c]              * ign_w[(3 * FIN + c) * FIN + f];
  for (int c = 0; c < FIN; ++c) s += tr[c]               * ign_w[(4 * FIN + c) * FIN + f];
  nb_t1[e * FIN + f] = s;
}

// coord_mlp: w[e] = SiLU(nb_t1 @ cw1 + cb1) @ cw2   (FIN->HID->1)
__global__ void k_coord_w(const float* __restrict__ nb_t1, const float* __restrict__ cw1,
                          const float* __restrict__ cb1, const float* __restrict__ cw2,
                          float* __restrict__ wgt) {
  __shared__ float red[HID];
  int e = blockIdx.x, t = threadIdx.x;  // 128 threads
  float a = cb1[t];
  for (int f = 0; f < FIN; ++f) a += nb_t1[e * FIN + f] * cw1[f * HID + t];
  red[t] = silu_f(a) * cw2[t];
  __syncthreads();
  for (int off = HID / 2; off > 0; off >>= 1) {
    if (t < off) red[t] += red[t + off];
    __syncthreads();
  }
  if (t == 0) wgt[e] = red[0];
}

__global__ void k_seg_scatter(const int* __restrict__ edges,
                              const float* __restrict__ cd,
                              const float* __restrict__ wgt, float* seg,
                              float* cnt) {
  int e = blockIdx.x * blockDim.x + threadIdx.x;
  if (e >= E_EDGES) return;
  int r = edges[e];
  float w = wgt[e];
  atomicAdd(&seg[r * 3 + 0], cd[e * 3 + 0] * w);
  atomicAdd(&seg[r * 3 + 1], cd[e * 3 + 1] * w);
  atomicAdd(&seg[r * 3 + 2], cd[e * 3 + 2] * w);
  atomicAdd(&cnt[r], 1.0f);
}

__global__ void k_xnew(const float* __restrict__ x, const float* __restrict__ seg,
                       const float* __restrict__ cnt, float* __restrict__ out_x) {
  int i = blockIdx.x * blockDim.x + threadIdx.x;
  if (i >= N_NODES * 3) return;
  int n = i / 3;
  float c = cnt[n];
  c = (c < 1.0f) ? 1.0f : c;
  out_x[i] = x[i] + seg[i] / c;
}

__global__ void k_nbt0_scatter(const int* __restrict__ edges,
                               const float* __restrict__ nb_t1,
                               float* __restrict__ nb_t0) {
  int i = blockIdx.x * blockDim.x + threadIdx.x;
  if (i >= E_EDGES * FIN) return;
  int e = i / FIN, f = i - e * FIN;
  atomicAdd(&nb_t0[edges[e] * FIN + f], nb_t1[i]);
}

// node_dec: h_new = h + SiLU(nb_t0 @ ndw1 + ndb1) @ ndw2 + ndb2
__global__ void k_hnew(const float* __restrict__ h, const float* __restrict__ nb_t0,
                       const float* __restrict__ ndw1, const float* __restrict__ ndb1,
                       const float* __restrict__ ndw2, const float* __restrict__ ndb2,
                       float* __restrict__ out_h) {
  __shared__ float hid[HID];
  int n = blockIdx.x, t = threadIdx.x;  // 128 threads
  float a = ndb1[t];
  for (int f = 0; f < FIN; ++f) a += nb_t0[n * FIN + f] * ndw1[f * HID + t];
  hid[t] = silu_f(a);
  __syncthreads();
  if (t < FIN) {
    float s = ndb2[t];
    for (int k = 0; k < HID; ++k) s += hid[k] * ndw2[k * FIN + t];
    out_h[n * FIN + t] = h[n * FIN + t] + s;
  }
}

// ---------------------------------------------------------------------------
// Workspace layout (bytes)
// ---------------------------------------------------------------------------
static constexpr size_t WS_T    = 0;                        // T f32 [R,96]; reused as Tout f32 [R,64]
static constexpr size_t WS_TB   = WS_T    + (size_t)R_TOTAL * 96 * 4;   // T bf16 [R,96]
static constexpr size_t WS_M1   = WS_TB   + (size_t)R_TOTAL * 96 * 2;   // m1 feat-major [64][i*512+k]
static constexpr size_t WS_M2   = WS_M1   + (size_t)R_TOTAL * 64 * 2;   // m2 feat-major TRANSPOSED [64][j*512+k]
static constexpr size_t WS_MULT = WS_M2   + (size_t)R_TOTAL * 64 * 2;   // mult feat-major [64][i*512+j]
static constexpr size_t WS_W    = WS_MULT + (size_t)R_TOTAL * 64 * 2;   // bf16 transposed weights
static constexpr size_t WB_EW1  = WS_W;
static constexpr size_t WB_EW2  = WB_EW1 + 16384 * 2;
static constexpr size_t WB_P1W1 = WB_EW2 + 8192 * 2;
static constexpr size_t WB_P1W2 = WB_P1W1 + 12288 * 2;
static constexpr size_t WB_P2W1 = WB_P1W2 + 8192 * 2;
static constexpr size_t WB_P2W2 = WB_P2W1 + 12288 * 2;
static constexpr size_t WB_P3W1 = WB_P2W2 + 8192 * 2;
static constexpr size_t WB_P3W2 = WB_P3W1 + 20480 * 2;
static constexpr size_t WS_XH   = WS_W + 262144;
static constexpr size_t WS_EFN  = WS_XH + 131072;
static constexpr size_t WS_CD   = WS_EFN + 65536;
static constexpr size_t WS_DIAG = WS_CD + 8192;
static constexpr size_t WS_ROW  = WS_DIAG + 131072;
static constexpr size_t WS_COL  = WS_ROW + 131072;
static constexpr size_t WS_TOT  = WS_COL + 131072;
static constexpr size_t WS_TR   = WS_TOT + 512;
static constexpr size_t WS_NBT1 = WS_TR + 512;
static constexpr size_t WS_WGT  = WS_NBT1 + 131072;
static constexpr size_t WS_SEG  = WS_WGT + 2048;
static constexpr size_t WS_CNT  = WS_SEG + 6144;
static constexpr size_t WS_NBT0 = WS_CNT + 2048;

static inline size_t mlp_shmem(int K1) {
  return (size_t)(K1 * HID + HID * FIN + 64 * K1 + 64 * HID) * sizeof(bf16);
}

extern "C" void kernel_launch(void* const* d_in, const int* in_sizes, int n_in,
                              void* d_out, int out_size, void* d_ws, size_t ws_size,
                              hipStream_t stream) {
  const float* h         = (const float*)d_in[0];
  const float* x         = (const float*)d_in[1];
  const int*   edges     = (const int*)d_in[2];
  const int*   nb_edge   = (const int*)d_in[3];
  const float* edge_attr = (const float*)d_in[4];
  const float* ew1 = (const float*)d_in[6];
  const float* eb1 = (const float*)d_in[7];
  const float* ew2 = (const float*)d_in[8];
  const float* eb2 = (const float*)d_in[9];
  const float* p1w1 = (const float*)d_in[10];
  const float* p1b1 = (const float*)d_in[11];
  const float* p1w2 = (const float*)d_in[12];
  const float* p1b2 = (const float*)d_in[13];
  const float* p2w1 = (const float*)d_in[14];
  const float* p2b1 = (const float*)d_in[15];
  const float* p2w2 = (const float*)d_in[16];
  const float* p2b2 = (const float*)d_in[17];
  const float* p3w1 = (const float*)d_in[18];
  const float* p3b1 = (const float*)d_in[19];
  const float* p3w2 = (const float*)d_in[20];
  const float* p3b2 = (const float*)d_in[21];
  const float* ign_w = (const float*)d_in[22];
  const float* ign_b = (const float*)d_in[23];
  const float* cw1 = (const float*)d_in[24];
  const float* cb1 = (const float*)d_in[25];
  const float* cw2 = (const float*)d_in[26];
  const float* ndw1 = (const float*)d_in[27];
  const float* ndb1 = (const float*)d_in[28];
  const float* ndw2 = (const float*)d_in[29];
  const float* ndb2 = (const float*)d_in[30];

  char* ws = (char*)d_ws;
  float* Tf    = (float*)(ws + WS_T);
  bf16*  Tb    = (bf16*)(ws + WS_TB);
  bf16*  m1t   = (bf16*)(ws + WS_M1);
  bf16*  m2t   = (bf16*)(ws + WS_M2);
  bf16*  mult  = (bf16*)(ws + WS_MULT);
  bf16*  ew1t  = (bf16*)(ws + WB_EW1);
  bf16*  ew2t  = (bf16*)(ws + WB_EW2);
  bf16*  p1w1t = (bf16*)(ws + WB_P1W1);
  bf16*  p1w2t = (bf16*)(ws + WB_P1W2);
  bf16*  p2w1t = (bf16*)(ws + WB_P2W1);
  bf16*  p2w2t = (bf16*)(ws + WB_P2W2);
  bf16*  p3w1t = (bf16*)(ws + WB_P3W1);
  bf16*  p3w2t = (bf16*)(ws + WB_P3W2);
  bf16*  xh    = (bf16*)(ws + WS_XH);
  bf16*  efn   = (bf16*)(ws + WS_EFN);
  float* cd    = (float*)(ws + WS_CD);
  float* diag  = (float*)(ws + WS_DIAG);
  float* rowsum = (float*)(ws + WS_ROW);
  float* colsum = (float*)(ws + WS_COL);
  float* tot   = (float*)(ws + WS_TOT);
  float* tr    = (float*)(ws + WS_TR);
  float* nbt1  = (float*)(ws + WS_NBT1);
  float* wgt   = (float*)(ws + WS_WGT);
  float* seg   = (float*)(ws + WS_SEG);
  float* cnt   = (float*)(ws + WS_CNT);
  float* nbt0  = (float*)(ws + WS_NBT0);
  float* Tout  = Tf;

  float* out_h = (float*)d_out;
  float* out_x = out_h + N_NODES * FIN;
  float* out_e = out_x + N_NODES * 3;

  // 1. transpose-convert weights to bf16 N-major
  k_f2b_t<<<64, 256, 0, stream>>>(ew1, ew1t, 128, 128);
  k_f2b_t<<<32, 256, 0, stream>>>(ew2, ew2t, 128, 64);
  k_f2b_t<<<48, 256, 0, stream>>>(p1w1, p1w1t, 96, 128);
  k_f2b_t<<<32, 256, 0, stream>>>(p1w2, p1w2t, 128, 64);
  k_f2b_t<<<48, 256, 0, stream>>>(p2w1, p2w1t, 96, 128);
  k_f2b_t<<<32, 256, 0, stream>>>(p2w2, p2w2t, 128, 64);
  k_f2b_t<<<80, 256, 0, stream>>>(p3w1, p3w1t, 160, 128);
  k_f2b_t<<<32, 256, 0, stream>>>(p3w2, p3w2t, 128, 64);

  // 2. edge prep
  k_prep<<<2, 256, 0, stream>>>(h, x, edges, cd, xh);

  // 3. efn = edge_encoder MLP (WMMA)
  k_fused_mlp<<<E_EDGES / 64, 128, mlp_shmem(128), stream>>>(
      xh, 128, nullptr, 0, ew1t, eb1, ew2t, eb2, E_EDGES, efn, nullptr, 0, nullptr);

  // 4. densify T
  k_zero_f32<<<8192, 256, 0, stream>>>(Tf, R_TOTAL * 96);
  k_vtv_scatter<<<M_NB, 96, 0, stream>>>(nb_edge, cd, efn, Tf);
  k_f2b<<<8192, 256, 0, stream>>>(Tf, Tb, R_TOTAL * 96);

  // 5. m1 (feature-major [i][k]) and m2 (feature-major transposed [j][k])
  k_fused_mlp<<<R_TOTAL / 64, 128, mlp_shmem(96), stream>>>(
      Tb, 96, nullptr, 0, p1w1t, p1b1, p1w2t, p1b2, R_TOTAL, nullptr, m1t, 0, nullptr);
  k_fused_mlp<<<R_TOTAL / 64, 128, mlp_shmem(96), stream>>>(
      Tb, 96, nullptr, 0, p2w1t, p2b1, p2w2t, p2b2, R_TOTAL, nullptr, m2t, 1, nullptr);

  // 6. feature-wise matmul: 64 x (512x512x512) bf16 GEMMs (WMMA)
  k_einsum<<<dim3(16, 64), 256, 0, stream>>>(m1t, m2t, mult);

  // 7. Tout = output MLP over concat(T, mult) (WMMA), fp32 out, reuses Tf buffer
  k_fused_mlp<<<R_TOTAL / 64, 128, mlp_shmem(160), stream>>>(
      Tb, 96, mult, 64, p3w1t, p3b1, p3w2t, p3b2, R_TOTAL, nullptr, nullptr, 0, Tout);

  // 8. IGN 2->1 reductions
  k_reduce_rows<<<EN, FIN, 0, stream>>>(Tout, rowsum, diag);
  k_reduce_cols<<<EN, FIN, 0, stream>>>(Tout, colsum);
  k_reduce_tot<<<1, FIN, 0, stream>>>(rowsum, diag, tot, tr);
  k_nb_t1<<<E_EDGES, FIN, 0, stream>>>(diag, rowsum, colsum, tot, tr, ign_w, ign_b, nbt1);

  // 9. coordinate update
  k_coord_w<<<E_EDGES, HID, 0, stream>>>(nbt1, cw1, cb1, cw2, wgt);
  k_zero_f32<<<8, 256, 0, stream>>>(seg, N_NODES * 3 + N_NODES);
  k_seg_scatter<<<2, 256, 0, stream>>>(edges, cd, wgt, seg, cnt);
  k_xnew<<<6, 256, 0, stream>>>(x, seg, cnt, out_x);

  // 10. node update
  k_zero_f32<<<128, 256, 0, stream>>>(nbt0, N_NODES * FIN);
  k_nbt0_scatter<<<128, 256, 0, stream>>>(edges, nbt1, nbt0);
  k_hnew<<<N_NODES, HID, 0, stream>>>(h, nbt0, ndw1, ndb1, ndw2, ndb2, out_h);

  // 11. pass-through edge_attr
  hipMemcpyAsync(out_e, edge_attr, (size_t)E_EDGES * 4 * sizeof(float),
                 hipMemcpyDeviceToDevice, stream);
}
